// TensorFactorizationMachineModel_58823872086140
// MI455X (gfx1250) — matrix operand order repaired and checked by reading, not compiled, not run on hardware
//
#include <hip/hip_runtime.h>

typedef __attribute__((ext_vector_type(16))) __bf16 v16bf;
typedef __attribute__((ext_vector_type(8)))  __bf16 v8bf;
typedef __attribute__((ext_vector_type(8)))  float  v8f;

#define NUM_FIELDS      39
#define PER_FIELD_VOCAB 10000
#define EMBED_DIM       128
#define KPAD            64      // 39 fields zero-padded to 64 (2 x K=32 WMMA steps)
#define NL              5       // W0 has L=2 slices, W1 has L=3 -> 5 stacked
#define RDIM            128
#define BATCH           2048

union FragU { v16bf v; v8bf h[2]; };

// ---------------------------------------------------------------------------
// Prep: convert W0/W1 (fp32, [l][r][f] with f contiguous) into one padded bf16
// tensor Wws[l][r][k], k in [0,64), zeros for k>=39.  Run once per launch.
// ---------------------------------------------------------------------------
__global__ void prep_w_kernel(const float* __restrict__ W0,
                              const float* __restrict__ W1,
                              __bf16* __restrict__ Wws) {
    const int total = NL * RDIM * KPAD;
    for (int i = blockIdx.x * blockDim.x + threadIdx.x; i < total;
         i += gridDim.x * blockDim.x) {
        int l = i / (RDIM * KPAD);
        int r = (i / KPAD) % RDIM;
        int k = i % KPAD;
        float v = 0.0f;
        if (k < NUM_FIELDS) {
            v = (l < 2) ? W0[(l * RDIM + r) * NUM_FIELDS + k]
                        : W1[((l - 2) * RDIM + r) * NUM_FIELDS + k];
        }
        Wws[i] = (__bf16)v;
    }
}

// ---------------------------------------------------------------------------
// Main: one workgroup (256 thr = 8 wave32) per batch sample.
//   wave w owns r-tile w; loops over 8 d-tiles; 5 WMMA accumulators per tile.
// ---------------------------------------------------------------------------
__global__ __launch_bounds__(256)
void tfm_kernel(const int*   __restrict__ x,
                const float* __restrict__ embed_table,
                const float* __restrict__ linear_table,
                const float* __restrict__ bias,
                const __bf16* __restrict__ Wws,
                float* __restrict__ out, int out_size) {
    __shared__ __align__(32) __bf16 embT[EMBED_DIM][KPAD];   // 16 KB, [d][f]
    __shared__ int   sidx[NUM_FIELDS];
    __shared__ float slin[NUM_FIELDS];
    __shared__ float sPart[256];

    const int tid = threadIdx.x;
    const int b   = blockIdx.x;

    if (tid < NUM_FIELDS) {
        int idx = x[b * NUM_FIELDS + tid] + tid * PER_FIELD_VOCAB;
        sidx[tid] = idx;
        slin[tid] = linear_table[idx];
    }
    // zero-fill embT (covers f = 39..63 padding)
    {
        unsigned int* p = (unsigned int*)&embT[0][0];
        for (int i = tid; i < EMBED_DIM * KPAD / 2; i += 256) p[i] = 0u;
    }
    __syncthreads();

    // gather fp32 embeddings -> bf16, transposed into LDS: embT[d][f]
    for (int i = tid; i < NUM_FIELDS * EMBED_DIM; i += 256) {
        int f = i >> 7;          // i / 128
        int d = i & 127;         // i % 128
        float v = embed_table[(long)sidx[f] * EMBED_DIM + d];
        embT[d][f] = (__bf16)v;
    }
    __syncthreads();

    const int lane = tid & 31;
    const int wave = tid >> 5;        // r-tile index 0..7
    const int m    = lane & 15;
    const int hi   = lane >> 4;       // half-wave select

    // Load all 10 A fragments (5 l-slices x 2 k-steps), keep register-resident.
    // A 16x32 bf16 layout: lane<16 -> M=lane, v0..3=K{kb+0..7}, v4..7=K{kb+16..23}
    //                      lane>=16 -> same M, K offset +8.
    FragU afrag[NL][2];
#pragma unroll
    for (int l = 0; l < NL; ++l) {
#pragma unroll
        for (int kk = 0; kk < 2; ++kk) {
            const __bf16* base =
                Wws + ((l * RDIM + wave * 16 + m) * KPAD) + kk * 32 + hi * 8;
            afrag[l][kk].h[0] = *(const v8bf*)(base);        // K = kb+koff+0..7
            afrag[l][kk].h[1] = *(const v8bf*)(base + 16);   // K = kb+koff+16..23
        }
    }

    const v8f vzero = {0.f, 0.f, 0.f, 0.f, 0.f, 0.f, 0.f, 0.f};
    float part = 0.0f;

    for (int dt = 0; dt < 8; ++dt) {
        // B 32x16 bf16 layout: lane<16 -> N=lane, v0..3=K{kb+0..7}, v4..7=K{kb+16..23};
        //                      lane>=16 -> same N, K offset +8.
        FragU bfrag[2];
        const int d = dt * 16 + m;
#pragma unroll
        for (int kk = 0; kk < 2; ++kk) {
            const __bf16* base = &embT[d][0] + kk * 32 + hi * 8;
            bfrag[kk].h[0] = *(const v8bf*)(base);           // K = kb+koff+0..7
            bfrag[kk].h[1] = *(const v8bf*)(base + 16);      // K = kb+koff+16..23
        }

        v8f acc[NL];
#pragma unroll
        for (int l = 0; l < NL; ++l) acc[l] = vzero;

#pragma unroll
        for (int kk = 0; kk < 2; ++kk) {
#pragma unroll
            for (int l = 0; l < NL; ++l) {
                acc[l] = __builtin_amdgcn_wmma_f32_16x16x32_bf16(
                    /*neg_a=*/false, afrag[l][kk].v,
                    /*neg_b=*/false, bfrag[kk].v,
                    /*c_mod=*/(short)0, acc[l],
                    /*reuse_a=*/false, /*reuse_b=*/false);
            }
        }

        // elementwise combine: prod over l, then sum tile entries
        v8f t = acc[0] * acc[1] + acc[2] * acc[3] * acc[4];
#pragma unroll
        for (int j = 0; j < 8; ++j) part += t[j];
    }

    // deterministic block reduction
    sPart[tid] = part;
    __syncthreads();
    for (int s = 128; s > 0; s >>= 1) {
        if (tid < s) sPart[tid] += sPart[tid + s];
        __syncthreads();
    }

    if (tid == 0) {
        float lin = bias[0];
        for (int f = 0; f < NUM_FIELDS; ++f) lin += slin[f];
        out[b] = lin + sPart[0];
        if (b == 0 && out_size > BATCH) out[BATCH] = 0.0f;  // tuple's trailing 0
    }
}

// ---------------------------------------------------------------------------
extern "C" void kernel_launch(void* const* d_in, const int* in_sizes, int n_in,
                              void* d_out, int out_size, void* d_ws, size_t ws_size,
                              hipStream_t stream) {
    const int*   x    = (const int*)d_in[0];
    const float* emb  = (const float*)d_in[1];
    const float* lin  = (const float*)d_in[2];
    const float* bias = (const float*)d_in[3];
    const float* W0   = (const float*)d_in[4];
    const float* W1   = (const float*)d_in[5];

    __bf16* Wws = (__bf16*)d_ws;   // NL*RDIM*KPAD bf16 = 80 KB scratch

    prep_w_kernel<<<20, 256, 0, stream>>>(W0, W1, Wws);
    tfm_kernel<<<BATCH, 256, 0, stream>>>(x, emb, lin, bias, Wws,
                                          (float*)d_out, out_size);
}
